// EdgeConv_SDP_85143431676078
// MI455X (gfx1250) — matrix operand order repaired and checked by reading, not compiled, not run on hardware
//
#include <hip/hip_runtime.h>
#include <hip/hip_bf16.h>

// ---------------------------------------------------------------------------
// EdgeConv GNN forward for MI455X (gfx1250, wave32, WMMA).
//
// All dense layers run as f16 WMMA GEMMs (v_wmma_f32_16x16x32_f16, f32 acc).
// Each wave computes a 16x(NT*16) output stripe; NT is a template parameter
// chosen so numNt % NT == 0 -> the inner loop is branch-free straight-line
// code: one A-fragment load + NT contiguous B-tile loads + NT WMMAs.
// Packed weights (~1.4 MB total) stay L2-resident; B tiles for one K-step
// are contiguous (kt-major packing) so they load with immediate offsets.
//
// Input flattening assumption (setup_inputs() dict insertion order,
// recursively): d_in[0]=node_feats(20000x32 f32), d_in[1]=edge_feats
// (320000x38 f32), d_in[2]=edge_index(2x320000 i32), then params leaves:
//   nmm1: bn0{g,b,m,v}=3..6  l0{W,b}=7,8  bn1=9..12  l1=13,14 bn2=15..18
//         l2=19,20 bn3=21..24
//   emm1: bn0=25..28 l0=29,30 bn1=31..34 l1=35,36 bn2=37..40 l2=41,42 bn3=43..46
//   nmm2: l0=47,48 bn0=49..52 l1=53,54 bn1=55..58 l2=59,60 bn2=61..64
//   emm2: bn0=65..68 l0=69,70 bn1=71..74 l1=75,76 bn2=77..80 l2=81,82 bn3=83..86
//   nhead: l0=87,88 l1=89,90 l2=91,92 l3=93,94
//   ehead: l0=95,96 l1=97,98 l2=99,100 l3=101,102 l4=103,104
// Output: d_out = [node_out(20000), edge_out(320000)] f32.
// ---------------------------------------------------------------------------

typedef __attribute__((ext_vector_type(16))) _Float16 v16h;
typedef __attribute__((ext_vector_type(8)))  _Float16 v8h;
typedef __attribute__((ext_vector_type(8)))  float    v8f;

union V16U { v16h v; v8h h[2]; };

#define N_EDGES 320000
#define N_NODES 20000
#define EPS_BN  1e-5f

// ------------------------- weight packing ---------------------------------
// Packed B layout, kt-major: element (k, n) with kt=k>>5, nt=n>>4,
// lane=(k&16)|(n&15), j=k&15 goes to
//   dst[(((kt*numNt)+nt)*32 + lane)*16 + j]
// so the numNt tiles of one K-step are a contiguous numNt*1024B block.
// Within a tile this uses contiguous k, the same order as the A loads (not the ISA table order)
// (lanes 0-15 hold K=0..15, lanes 16-31 hold K=16..31, col = lane&15).
__global__ void pack_w_kernel(const float* __restrict__ W, _Float16* __restrict__ dst,
                              int K, int N, int Kpad, int Npad) {
    int i = blockIdx.x * blockDim.x + threadIdx.x;
    int total = Kpad * Npad;
    if (i >= total) return;
    int k = i / Npad, n = i % Npad;
    float val = (k < K && n < N) ? W[(size_t)k * N + n] : 0.0f;
    int kt = k >> 5, nt = n >> 4;
    int lane = (k & 16) | (n & 15);
    int j = k & 15;
    int numNt = Npad >> 4;
    dst[(((size_t)kt * numNt + nt) * 32 + lane) * 16 + j] = (_Float16)val;
}

// --------------------------- WMMA GEMM ------------------------------------
// Out[M x Npad](f16) = epilogue(A[M x Kpad](f16) @ Bpacked + bias)
// epilogue: optional relu, optional BN affine (computed from g,b,m,v).
// One wave computes a 16-row x (NT*16)-col stripe; block = 4 waves = 4
// consecutive row tiles.  Requires M % 16 == 0 and numNt % NT == 0 (true for
// every call here) -> zero guards in the inner loop.
template <int NT>
__global__ __launch_bounds__(128) void gemm_wmma_kernel(
    const _Float16* __restrict__ A, const _Float16* __restrict__ Bp,
    const float* __restrict__ bias, _Float16* __restrict__ Out,
    int M, int N, int Kpad, int Npad, int relu,
    const float* __restrict__ bng, const float* __restrict__ bnb,
    const float* __restrict__ bnm, const float* __restrict__ bnv) {
    const int lane = threadIdx.x & 31;
    const int wave = threadIdx.x >> 5;
    const int rowBase = (blockIdx.y * 4 + wave) * 16;
    if (rowBase >= M) return;                 // wave-uniform exit, EXEC stays full
    const int numKt = Kpad >> 5;
    const int numNt = Npad >> 4;
    const int ntBase = blockIdx.x * NT;

    const int halfSel = lane >> 4;            // 0: lanes 0-15, 1: lanes 16-31
    const int r0 = lane & 15;

    v8f acc[NT] = {};
    const _Float16* Arow = A + (size_t)(rowBase + r0) * Kpad + halfSel * 16;
    // B base for this wave's column group at kt=0; per-kt stride numNt*512 halfs.
    const _Float16* Bg = Bp + ((size_t)ntBase * 32 + lane) * 16;

    for (int kt = 0; kt < numKt; ++kt) {
        V16U a;
        const v8h* pa = (const v8h*)(Arow + kt * 32);
        a.h[0] = pa[0];                       // K = kb .. kb+7
        a.h[1] = pa[1];                       // K = kb+8 .. kb+15
        const _Float16* bptr = Bg + (size_t)kt * numNt * 512;
        #pragma unroll
        for (int t = 0; t < NT; ++t) {
            const v8h* pb = (const v8h*)(bptr + t * 512);  // immediate offsets
            V16U b;
            b.h[0] = pb[0];
            b.h[1] = pb[1];
            acc[t] = __builtin_amdgcn_wmma_f32_16x16x32_f16(
                /*neg_a=*/false, a.v, /*neg_b=*/false, b.v,
                /*c_mod=*/(short)0, acc[t], /*reuse_a=*/false, /*reuse_b=*/false);
        }
    }

    // epilogue (per-lane column, per tile)
    const bool aff = (bng != nullptr);
    #pragma unroll
    for (int t = 0; t < NT; ++t) {
        const int col = (ntBase + t) * 16 + r0;
        const bool inN = (col < N);
        float bv = (inN && bias) ? bias[col] : 0.0f;
        float s = 1.0f, tt = 0.0f;
        if (aff && inN) {
            float rs = rsqrtf(bnv[col] + EPS_BN);
            s = bng[col] * rs;
            tt = bnb[col] - bnm[col] * s;
        }
        #pragma unroll
        for (int r = 0; r < 8; ++r) {
            int row = rowBase + r + halfSel * 8;
            float y = acc[t][r] + bv;
            if (relu) y = fmaxf(y, 0.0f);
            if (aff)  y = inN ? (y * s + tt) : 0.0f;
            Out[(size_t)row * Npad + col] = (_Float16)y;
        }
    }
}

// ----------------------- prep / gather kernels ----------------------------
__global__ void prep_cat1_kernel(const float* __restrict__ nf,
                                 const int* __restrict__ src, const int* __restrict__ dst,
                                 const float* __restrict__ g, const float* __restrict__ b,
                                 const float* __restrict__ m, const float* __restrict__ v,
                                 _Float16* __restrict__ out) {
    long i = (long)blockIdx.x * blockDim.x + threadIdx.x;
    if (i >= (long)N_EDGES * 32) return;
    int e = (int)(i >> 5), k = (int)(i & 31);
    int sn = src[e], dn = dst[e];
    float val;
    if (k < 16) val = nf[(size_t)dn * 32 + k];
    else { int kk = k - 16; val = nf[(size_t)sn * 32 + kk] - nf[(size_t)dn * 32 + kk]; }
    float rs = rsqrtf(v[k] + EPS_BN);
    float s = g[k] * rs;
    out[i] = (_Float16)(val * s + (b[k] - m[k] * s));
}

__global__ void prep_cate1_kernel(const float* __restrict__ ef,
                                  const int* __restrict__ src, const int* __restrict__ dst,
                                  const float* __restrict__ x1,
                                  const float* __restrict__ g, const float* __restrict__ b,
                                  const float* __restrict__ m, const float* __restrict__ v,
                                  _Float16* __restrict__ out) {
    long i = (long)blockIdx.x * blockDim.x + threadIdx.x;
    if (i >= (long)N_EDGES * 160) return;
    int e = (int)(i / 160), k = (int)(i % 160);
    float val = 0.0f;
    if (k < 19)       val = ef[(size_t)e * 38 + k];
    else if (k < 83)  val = x1[(size_t)src[e] * 64 + (k - 19)];
    else if (k < 147) val = x1[(size_t)dst[e] * 64 + (k - 83)];
    if (k < 147) {
        float rs = rsqrtf(v[k] + EPS_BN);
        float s = g[k] * rs;
        val = val * s + (b[k] - m[k] * s);
    }
    out[i] = (_Float16)val;
}

__global__ void prep_cat2_kernel(const int* __restrict__ src, const int* __restrict__ dst,
                                 const float* __restrict__ x1, _Float16* __restrict__ out) {
    long i = (long)blockIdx.x * blockDim.x + threadIdx.x;
    if (i >= (long)N_EDGES * 128) return;
    int e = (int)(i >> 7), k = (int)(i & 127);
    int sn = src[e], dn = dst[e];
    float val;
    if (k < 64) val = x1[(size_t)dn * 64 + k];
    else { int kk = k - 64; val = x1[(size_t)sn * 64 + kk] - x1[(size_t)dn * 64 + kk]; }
    out[i] = (_Float16)val;
}

__global__ void prep_cat3_kernel(const _Float16* __restrict__ e1k,
                                 const int* __restrict__ src, const int* __restrict__ dst,
                                 const float* __restrict__ x2,
                                 const float* __restrict__ g, const float* __restrict__ b,
                                 const float* __restrict__ m, const float* __restrict__ v,
                                 _Float16* __restrict__ out) {
    long i = (long)blockIdx.x * blockDim.x + threadIdx.x;
    if (i >= (long)N_EDGES * 352) return;
    int e = (int)(i / 352), k = (int)(i % 352);
    float val = 0.0f;
    if (k < 76)       val = (float)e1k[(size_t)e * 96 + k];
    else if (k < 204) val = x2[(size_t)src[e] * 128 + (k - 76)];
    else if (k < 332) val = x2[(size_t)dst[e] * 128 + (k - 204)];
    if (k < 332) {
        float rs = rsqrtf(v[k] + EPS_BN);
        float s = g[k] * rs;
        val = val * s + (b[k] - m[k] * s);
    }
    out[i] = (_Float16)val;
}

__global__ void f32_to_f16_kernel(const float* __restrict__ in, _Float16* __restrict__ out, long n) {
    long i = (long)blockIdx.x * blockDim.x + threadIdx.x;
    if (i < n) out[i] = (_Float16)in[i];
}

__global__ void zero_f32_kernel(float* __restrict__ p, long n) {
    long i = (long)blockIdx.x * blockDim.x + threadIdx.x;
    if (i < n) p[i] = 0.0f;
}

// segment_sum: x[dst[e]] += m[e]   (f16 source, f32 accumulator)
__global__ void scatter_add_kernel(const _Float16* __restrict__ Mbuf,
                                   const int* __restrict__ dst,
                                   float* __restrict__ X, int W) {
    long i = (long)blockIdx.x * blockDim.x + threadIdx.x;
    if (i >= (long)N_EDGES * W) return;
    int e = (int)(i / W), f = (int)(i % W);
    atomicAdd(&X[(size_t)dst[e] * W + f], (float)Mbuf[(size_t)e * W + f]);
}

// final N=1 layer: out = sigmoid(row . W + b), one wave per row
__global__ __launch_bounds__(128) void dot_sigmoid_kernel(
    const _Float16* __restrict__ A, const float* __restrict__ W,
    const float* __restrict__ bias, float* __restrict__ out,
    int M, int K, int Kpad) {
    int lane = threadIdx.x & 31;
    int row = blockIdx.x * 4 + (threadIdx.x >> 5);
    if (row >= M) return;
    const _Float16* a = A + (size_t)row * Kpad;
    float s = 0.0f;
    for (int k = lane; k < K; k += 32) s += (float)a[k] * W[k];
    #pragma unroll
    for (int off = 16; off; off >>= 1) s += __shfl_xor(s, off, 32);
    if (lane == 0) out[row] = 1.0f / (1.0f + expf(-(s + bias[0])));
}

// ------------------------------ host side ---------------------------------
struct Layer { int wi, bi, K, N, Kp, Np, relu, bn; };

extern "C" void kernel_launch(void* const* d_in, const int* in_sizes, int n_in,
                              void* d_out, int out_size, void* d_ws, size_t ws_size,
                              hipStream_t stream) {
    (void)in_sizes; (void)n_in; (void)out_size;
    auto F = [&](int i) { return (const float*)d_in[i]; };
    const float* node_feats = F(0);
    const float* edge_feats = F(1);
    const int*   src = (const int*)d_in[2];
    const int*   dst = src + N_EDGES;

    static const Layer Ls[19] = {
        {  7,   8,  32,  64,  32,  64, 1,  9},   //  0 nmm1.l0  (+bn1)
        { 13,  14,  64,  64,  64,  64, 1, 15},   //  1 nmm1.l1  (+bn2)
        { 19,  20,  64,  64,  64,  64, 1, 21},   //  2 nmm1.l2  (+bn3)
        { 29,  30, 147,  76, 160,  96, 1, 31},   //  3 emm1.l0  (+bn1)
        { 35,  36,  76,  76,  96,  96, 1, 37},   //  4 emm1.l1  (+bn2)
        { 41,  42,  76,  76,  96,  96, 1, 43},   //  5 emm1.l2  (+bn3)
        { 47,  48, 128, 128, 128, 128, 1, 49},   //  6 nmm2.l0  (+bn0)
        { 53,  54, 128, 128, 128, 128, 1, 55},   //  7 nmm2.l1  (+bn1)
        { 59,  60, 128, 128, 128, 128, 1, 61},   //  8 nmm2.l2  (+bn2)
        { 69,  70, 332, 152, 352, 160, 1, 71},   //  9 emm2.l0  (+bn1)
        { 75,  76, 152, 152, 160, 160, 1, 77},   // 10 emm2.l1  (+bn2)
        { 81,  82, 152, 152, 160, 160, 1, 83},   // 11 emm2.l2  (+bn3)
        { 87,  88, 128, 256, 128, 256, 1, -1},   // 12 nhead.l0
        { 89,  90, 256, 256, 256, 256, 1, -1},   // 13 nhead.l1
        { 91,  92, 256, 256, 256, 256, 0, -1},   // 14 nhead.l2
        { 95,  96, 152, 304, 160, 320, 0, -1},   // 15 ehead.l0 (no relu)
        { 97,  98, 304, 304, 320, 320, 1, -1},   // 16 ehead.l1
        { 99, 100, 304, 304, 320, 320, 1, -1},   // 17 ehead.l2
        {101, 102, 304, 304, 320, 320, 0, -1},   // 18 ehead.l3
    };

    // packed-weight offsets (halfs)
    size_t pwOff[19], pwTot = 0;
    for (int i = 0; i < 19; ++i) { pwOff[i] = pwTot; pwTot += (size_t)Ls[i].Kp * Ls[i].Np; }

    // workspace bump allocation (bytes, 256-aligned)
    size_t off = 0;
    auto alloc = [&](size_t bytes) { size_t o = off; off = (off + bytes + 255) & ~(size_t)255; return o; };
    size_t oPW  = alloc(pwTot * 2);
    size_t oX1  = alloc((size_t)N_NODES * 64 * 4);
    size_t oX2  = alloc((size_t)N_NODES * 128 * 4);
    size_t oX2H = alloc((size_t)N_NODES * 128 * 2);
    size_t oPN0 = alloc((size_t)N_NODES * 256 * 2);
    size_t oPN1 = alloc((size_t)N_NODES * 256 * 2);
    size_t oE1K = alloc((size_t)N_EDGES * 96 * 2);
    size_t oPB0 = alloc((size_t)N_EDGES * 352 * 2);
    size_t oPB1 = alloc((size_t)N_EDGES * 320 * 2);
    size_t oPB2 = alloc((size_t)N_EDGES * 320 * 2);
    if (off > ws_size) return;   // workspace too small: bail (no runtime here)

    char* ws = (char*)d_ws;
    _Float16* PW  = (_Float16*)(ws + oPW);
    float*    X1  = (float*)(ws + oX1);
    float*    X2  = (float*)(ws + oX2);
    _Float16* X2H = (_Float16*)(ws + oX2H);
    _Float16* PN0 = (_Float16*)(ws + oPN0);
    _Float16* PN1 = (_Float16*)(ws + oPN1);
    _Float16* E1K = (_Float16*)(ws + oE1K);
    _Float16* PB0 = (_Float16*)(ws + oPB0);
    _Float16* PB1 = (_Float16*)(ws + oPB1);
    _Float16* PB2 = (_Float16*)(ws + oPB2);

    // 1) pack all weights into WMMA B layout (f16, zero-padded, kt-major)
    for (int i = 0; i < 19; ++i) {
        int total = Ls[i].Kp * Ls[i].Np;
        pack_w_kernel<<<(total + 255) / 256, 256, 0, stream>>>(
            F(Ls[i].wi), PW + pwOff[i], Ls[i].K, Ls[i].N, Ls[i].Kp, Ls[i].Np);
    }

    // 2) zero node accumulators
    {
        long n1 = (long)N_NODES * 64, n2 = (long)N_NODES * 128;
        zero_f32_kernel<<<(int)((n1 + 255) / 256), 256, 0, stream>>>(X1, n1);
        zero_f32_kernel<<<(int)((n2 + 255) / 256), 256, 0, stream>>>(X2, n2);
    }

    auto gemm = [&](const _Float16* A, int li, _Float16* Out, int M) {
        const Layer& l = Ls[li];
        int numNt = l.Np >> 4;
        const float* g = (l.bn >= 0) ? F(l.bn)     : nullptr;
        const float* b = (l.bn >= 0) ? F(l.bn + 1) : nullptr;
        const float* m = (l.bn >= 0) ? F(l.bn + 2) : nullptr;
        const float* v = (l.bn >= 0) ? F(l.bn + 3) : nullptr;
        dim3 blk(128);
        int gy = (M + 63) / 64;
        // pick largest NT in {10,8,6,4} dividing numNt (covers 4,6,8,10,16,20)
        if (numNt % 10 == 0) {
            gemm_wmma_kernel<10><<<dim3(numNt / 10, gy), blk, 0, stream>>>(
                A, PW + pwOff[li], F(l.bi), Out, M, l.N, l.Kp, l.Np, l.relu, g, b, m, v);
        } else if (numNt % 8 == 0) {
            gemm_wmma_kernel<8><<<dim3(numNt / 8, gy), blk, 0, stream>>>(
                A, PW + pwOff[li], F(l.bi), Out, M, l.N, l.Kp, l.Np, l.relu, g, b, m, v);
        } else if (numNt % 6 == 0) {
            gemm_wmma_kernel<6><<<dim3(numNt / 6, gy), blk, 0, stream>>>(
                A, PW + pwOff[li], F(l.bi), Out, M, l.N, l.Kp, l.Np, l.relu, g, b, m, v);
        } else {
            gemm_wmma_kernel<4><<<dim3(numNt / 4, gy), blk, 0, stream>>>(
                A, PW + pwOff[li], F(l.bi), Out, M, l.N, l.Kp, l.Np, l.relu, g, b, m, v);
        }
    };
    auto blocksFor = [](long n) { return (int)((n + 255) / 256); };

    // ---- nmm1: cat1 -> 64-wide messages -> segment_sum into X1 ----
    prep_cat1_kernel<<<blocksFor((long)N_EDGES * 32), 256, 0, stream>>>(
        node_feats, src, dst, F(3), F(4), F(5), F(6), PB0);
    gemm(PB0, 0, PB1, N_EDGES);
    gemm(PB1, 1, PB2, N_EDGES);
    gemm(PB2, 2, PB0, N_EDGES);                       // m1 (stride 64)
    scatter_add_kernel<<<blocksFor((long)N_EDGES * 64), 256, 0, stream>>>(PB0, dst, X1, 64);

    // ---- emm1: [e, x1[src], x1[dst]] -> e1 (kept, stride 96) ----
    prep_cate1_kernel<<<blocksFor((long)N_EDGES * 160), 256, 0, stream>>>(
        edge_feats, src, dst, X1, F(25), F(26), F(27), F(28), PB1);
    gemm(PB1, 3, PB2, N_EDGES);
    gemm(PB2, 4, PB0, N_EDGES);
    gemm(PB0, 5, E1K, N_EDGES);

    // ---- nmm2: [xi, xj-xi] (x=X1) -> 128-wide messages -> X2 ----
    prep_cat2_kernel<<<blocksFor((long)N_EDGES * 128), 256, 0, stream>>>(src, dst, X1, PB0);
    gemm(PB0, 6, PB1, N_EDGES);
    gemm(PB1, 7, PB2, N_EDGES);
    gemm(PB2, 8, PB0, N_EDGES);                       // m2 (stride 128)
    scatter_add_kernel<<<blocksFor((long)N_EDGES * 128), 256, 0, stream>>>(PB0, dst, X2, 128);

    // ---- emm2: [e1, x2[src], x2[dst]] -> e2 (stride 160, ends in PB1) ----
    prep_cat3_kernel<<<blocksFor((long)N_EDGES * 352), 256, 0, stream>>>(
        E1K, src, dst, X2, F(65), F(66), F(67), F(68), PB0);
    gemm(PB0,  9, PB1, N_EDGES);
    gemm(PB1, 10, PB2, N_EDGES);
    gemm(PB2, 11, PB1, N_EDGES);                      // e2 final

    // ---- node head: X2 -> sigmoid scalar ----
    {
        long n = (long)N_NODES * 128;
        f32_to_f16_kernel<<<blocksFor(n), 256, 0, stream>>>(X2, X2H, n);
    }
    gemm(X2H, 12, PN0, N_NODES);
    gemm(PN0, 13, PN1, N_NODES);
    gemm(PN1, 14, PN0, N_NODES);
    dot_sigmoid_kernel<<<(N_NODES + 3) / 4, 128, 0, stream>>>(
        PN0, F(93), F(94), (float*)d_out, N_NODES, 256, 256);

    // ---- edge head: e2 -> sigmoid scalar ----
    gemm(PB1, 15, PB2, N_EDGES);
    gemm(PB2, 16, PB0, N_EDGES);
    gemm(PB0, 17, PB2, N_EDGES);
    gemm(PB2, 18, PB0, N_EDGES);
    dot_sigmoid_kernel<<<(N_EDGES + 3) / 4, 128, 0, stream>>>(
        PB0, F(103), F(104), (float*)d_out + N_NODES, N_EDGES, 304, 320);
}
